// GlobalEncoder_63866163692314
// MI455X (gfx1250) — matrix operand order repaired and checked by reading, not compiled, not run on hardware
//
#include <hip/hip_runtime.h>
#include <hip/hip_bf16.h>
#include <math.h>

#define N_TOK 4096
#define DIM   512
#define NHEAD 16
#define DHEAD 32
#define FF    2048
#define NDEPTH 3
#define NGENES 250
#define GRH   64
#define GRW   64

typedef __attribute__((ext_vector_type(16))) _Float16 v16h;
typedef __attribute__((ext_vector_type(8)))  _Float16 v8h;
typedef __attribute__((ext_vector_type(8)))  float    v8f;

// ---------------- WMMA fragment loaders (contiguous k = 16h + s order, same for A and B) ----------------
// A-matrix 16x32 f16: lane L holds row L%16; halves K = {0..15} (L<16) or {16..31}
static __device__ __forceinline__ v16h load_a_frag(const _Float16* __restrict__ A, int lda,
                                                   int m0, int k0, int lane) {
  const int row = m0 + (lane & 15);
  const int kb  = k0 + ((lane & 16) ? 16 : 0);
  const _Float16* p = A + (size_t)row * lda + kb;
  v8h lo = *(const v8h*)p;
  v8h hi = *(const v8h*)(p + 8);
  v16h r;
#pragma unroll
  for (int i = 0; i < 8; ++i) { r[i] = lo[i]; r[i + 8] = hi[i]; }
  return r;
}

// B-matrix 32x16 f16 where B[k][n] = W[n][k], W row-major (Nout x K):
// lane L holds column n = L%16; halves K = {0..15} (L<16) or {16..31} -> one contiguous 32B read
static __device__ __forceinline__ v16h load_b_frag(const _Float16* __restrict__ W, int ldw,
                                                   int n0, int k0, int lane, int nmax) {
  int col = n0 + (lane & 15);
  if (col >= nmax) col = nmax - 1;   // clamp (tail tiles); store side is guarded
  const int kb = k0 + ((lane & 16) ? 16 : 0);
  return *(const v16h*)(W + (size_t)col * ldw + kb);
}

static __device__ __forceinline__ float gelu_exact(float x) {
  return 0.5f * x * (1.0f + erff(x * 0.70710678118654752f));
}

// ---------------- Generic GEMM: C[M x Nc] = A[M x K] @ W[Nc x K]^T + bias ----------------
// block = 128 threads (4 waves); block tile = 128 rows x 64 cols.
// Each wave computes 32x64 (2 A-frags x 4 B-frags = 8 WMMAs per 32-wide K-step).
// The shared 64x32 B tile is staged through LDS once per block, double-buffered so the
// next tile's global loads overlap the current tile's WMMAs.
__global__ __launch_bounds__(128) void gemm_wmma_kernel(
    const _Float16* __restrict__ A, const _Float16* __restrict__ W,
    const float* __restrict__ bias, const float* __restrict__ res,
    float* __restrict__ outF, _Float16* __restrict__ outH,
    int M, int Nc, int K, int fuse_gelu)
{
  (void)M;
  __shared__ __align__(32) _Float16 btile[2][64 * 32];
  const int lane = threadIdx.x & 31;
  const int wave = threadIdx.x >> 5;
  const int m0 = blockIdx.y * 128 + wave * 32;
  const int n0 = blockIdx.x * 64;

  // cooperative B staging: thread t moves 32B: row t/2 of the tile, 16-half chunk (t&1)
  const int trow = threadIdx.x >> 1;
  const int tcol = (threadIdx.x & 1) * 16;
  int gcol = n0 + trow;
  if (gcol >= Nc) gcol = Nc - 1;           // clamp tail; store side guarded
  const _Float16* wrow = W + (size_t)gcol * K + tcol;

  // stage K-chunk 0
  *(v16h*)(&btile[0][trow * 32 + tcol]) = *(const v16h*)wrow;

  v8f acc0[4] = {}, acc1[4] = {};
  const int nk = K >> 5;
  int cur = 0;
  for (int kt = 0; kt < nk; ++kt) {
    __syncthreads();                        // btile[cur] ready for all waves
    const int k0 = kt << 5;
    v16h nxt;
    const bool hn = (kt + 1) < nk;
    if (hn) nxt = *(const v16h*)(wrow + k0 + 32);   // prefetch next B chunk to regs

    const v16h a0 = load_a_frag(A, K, m0,      k0, lane);
    const v16h a1 = load_a_frag(A, K, m0 + 16, k0, lane);
#pragma unroll
    for (int t = 0; t < 4; ++t) {
      const v16h b = *(const v16h*)(&btile[cur][(t * 16 + (lane & 15)) * 32 +
                                                ((lane & 16) ? 16 : 0)]);
      acc0[t] = __builtin_amdgcn_wmma_f32_16x16x32_f16(
          false, a0, false, b, (short)0, acc0[t], false, false);
      acc1[t] = __builtin_amdgcn_wmma_f32_16x16x32_f16(
          false, a1, false, b, (short)0, acc1[t], false, false);
    }
    if (hn) {
      *(v16h*)(&btile[cur ^ 1][trow * 32 + tcol]) = nxt;
      cur ^= 1;
    }
  }

  const int cl   = lane & 15;
  const int rofs = (lane & 16) ? 8 : 0;
#pragma unroll
  for (int t = 0; t < 4; ++t) {
    const int col = n0 + t * 16 + cl;
    if (col >= Nc) continue;
    const float bv = bias ? bias[col] : 0.0f;
#pragma unroll
    for (int r = 0; r < 8; ++r) {
      {
        const int row = m0 + rofs + r;
        float v = acc0[t][r] + bv;
        if (fuse_gelu) v = gelu_exact(v);
        const size_t idx = (size_t)row * Nc + col;
        if (res)  v += res[idx];
        if (outF) outF[idx] = v;
        if (outH) outH[idx] = (_Float16)v;
      }
      {
        const int row = m0 + 16 + rofs + r;
        float v = acc1[t][r] + bv;
        if (fuse_gelu) v = gelu_exact(v);
        const size_t idx = (size_t)row * Nc + col;
        if (res)  v += res[idx];
        if (outF) outF[idx] = v;
        if (outH) outH[idx] = (_Float16)v;
      }
    }
  }
}

// ---------------- LayerNorm (one block per row, 256 threads, D=512) ----------------
__global__ __launch_bounds__(256) void layernorm_kernel(
    const float* __restrict__ x, const float* __restrict__ g, const float* __restrict__ b,
    _Float16* __restrict__ outH, float* __restrict__ outF)
{
  __shared__ float red[256];
  const int row = blockIdx.x;
  const int tid = threadIdx.x;
  const float* xr = x + (size_t)row * DIM;
  const float v0 = xr[tid], v1 = xr[tid + 256];

  red[tid] = v0 + v1;
  __syncthreads();
  for (int ofs = 128; ofs > 0; ofs >>= 1) {
    if (tid < ofs) red[tid] += red[tid + ofs];
    __syncthreads();
  }
  const float mean = red[0] * (1.0f / DIM);
  __syncthreads();

  const float d0 = v0 - mean, d1 = v1 - mean;
  red[tid] = d0 * d0 + d1 * d1;
  __syncthreads();
  for (int ofs = 128; ofs > 0; ofs >>= 1) {
    if (tid < ofs) red[tid] += red[tid + ofs];
    __syncthreads();
  }
  const float inv = rsqrtf(red[0] * (1.0f / DIM) + 1e-5f);

  const float y0 = d0 * inv * g[tid]       + b[tid];
  const float y1 = d1 * inv * g[tid + 256] + b[tid + 256];
  outH[(size_t)row * DIM + tid]       = (_Float16)y0;
  outH[(size_t)row * DIM + tid + 256] = (_Float16)y1;
  if (outF) {
    outF[(size_t)row * DIM + tid]       = y0;
    outF[(size_t)row * DIM + tid + 256] = y1;
  }
}

// ---------------- QKV repack: (N,3D) f16 -> Q/K (nh,N,dh), V^T (nh,dh,N); scale folded into Q ---
__global__ void qkv_pack_kernel(const _Float16* __restrict__ qkv,
                                _Float16* __restrict__ Qh, _Float16* __restrict__ Kh,
                                _Float16* __restrict__ Vt)
{
  const int idx = blockIdx.x * blockDim.x + threadIdx.x;
  if (idx >= N_TOK * DIM) return;
  const int n = idx / DIM, c = idx % DIM;
  const int h = c / DHEAD, d = c % DHEAD;
  const size_t base = (size_t)n * (3 * DIM);
  const float qv = (float)qkv[base + c];
  const _Float16 kv = qkv[base + DIM + c];
  const _Float16 vv = qkv[base + 2 * DIM + c];
  Qh[((size_t)h * N_TOK + n) * DHEAD + d] = (_Float16)(qv * 0.17677669529663687f); // 1/sqrt(32)
  Kh[((size_t)h * N_TOK + n) * DHEAD + d] = kv;
  Vt[((size_t)h * DHEAD + d) * N_TOK + n] = vv;
}

// ---------------- Flash attention: one wave per (head, 16-query tile) ----------------
__global__ __launch_bounds__(128) void attention_kernel(
    const _Float16* __restrict__ Qh, const _Float16* __restrict__ Kh,
    const _Float16* __restrict__ Vt, _Float16* __restrict__ aoH)
{
  __shared__ __align__(32) _Float16 pbuf[4][16 * 32];
  const int lane = threadIdx.x & 31;
  const int wave = threadIdx.x >> 5;
  const int h  = blockIdx.y;
  const int q0 = blockIdx.x * 64 + wave * 16;

  const _Float16* Qb = Qh + (size_t)h * N_TOK * DHEAD;
  const _Float16* Kb = Kh + (size_t)h * N_TOK * DHEAD;
  const _Float16* Vb = Vt + (size_t)h * DHEAD * N_TOK;

  const v16h qf = load_a_frag(Qb, DHEAD, q0, 0, lane);
  v8f accO[2] = {};
  float mrow[8], lrow[8];
#pragma unroll
  for (int r = 0; r < 8; ++r) { mrow[r] = -1e30f; lrow[r] = 0.0f; }
  const int cl   = lane & 15;
  const int rofs = (lane & 16) ? 8 : 0;

  for (int kb = 0; kb < N_TOK; kb += 32) {
    // scores for 32 keys: two 16x16 tiles, K-dim = dh = 32 (one WMMA each)
    const v16h kf0 = load_b_frag(Kb, DHEAD, kb,      0, lane, N_TOK);
    const v16h kf1 = load_b_frag(Kb, DHEAD, kb + 16, 0, lane, N_TOK);
    v8f s0 = {}, s1 = {};
    s0 = __builtin_amdgcn_wmma_f32_16x16x32_f16(false, qf, false, kf0, (short)0, s0, false, false);
    s1 = __builtin_amdgcn_wmma_f32_16x16x32_f16(false, qf, false, kf1, (short)0, s1, false, false);

    // online softmax: per-row stats via half-wave reductions (rows live in the correct half)
#pragma unroll
    for (int r = 0; r < 8; ++r) {
      float smax = fmaxf(s0[r], s1[r]);
#pragma unroll
      for (int m = 8; m >= 1; m >>= 1) smax = fmaxf(smax, __shfl_xor(smax, m, 32));
      const float mnew  = fmaxf(mrow[r], smax);
      const float alpha = __expf(mrow[r] - mnew);
      const float e0 = __expf(s0[r] - mnew);
      const float e1 = __expf(s1[r] - mnew);
      float esum = e0 + e1;
#pragma unroll
      for (int m = 8; m >= 1; m >>= 1) esum += __shfl_xor(esum, m, 32);
      lrow[r] = lrow[r] * alpha + esum;
      mrow[r] = mnew;
      accO[0][r] *= alpha;
      accO[1][r] *= alpha;
      // transpose P (D-frag layout) into LDS so we can reload it as an A-fragment
      pbuf[wave][(rofs + r) * 32 + cl]      = (_Float16)e0;
      pbuf[wave][(rofs + r) * 32 + 16 + cl] = (_Float16)e1;
    }

    const v16h pf = load_a_frag(&pbuf[wave][0], 32, 0, 0, lane);
    const int kofs = kb + ((lane & 16) ? 16 : 0);
    const v16h vf0 = *(const v16h*)(Vb + (size_t)cl        * N_TOK + kofs);
    const v16h vf1 = *(const v16h*)(Vb + (size_t)(16 + cl) * N_TOK + kofs);
    accO[0] = __builtin_amdgcn_wmma_f32_16x16x32_f16(false, pf, false, vf0, (short)0, accO[0], false, false);
    accO[1] = __builtin_amdgcn_wmma_f32_16x16x32_f16(false, pf, false, vf1, (short)0, accO[1], false, false);
  }

#pragma unroll
  for (int t = 0; t < 2; ++t)
#pragma unroll
    for (int r = 0; r < 8; ++r)
      aoH[(size_t)(q0 + rofs + r) * DIM + h * DHEAD + t * 16 + cl] =
          (_Float16)(accO[t][r] / lrow[r]);
}

// ---------------- Depthwise 3x3 conv over scattered 64x64 grid ----------------
__global__ void grid_init_kernel(int* __restrict__ map) {
  const int idx = blockIdx.x * blockDim.x + threadIdx.x;
  if (idx < GRH * GRW) map[idx] = -1;
}
__global__ void grid_scatter_kernel(const int* __restrict__ coords, int* __restrict__ map) {
  const int n = blockIdx.x * blockDim.x + threadIdx.x;
  if (n < N_TOK) map[coords[2 * n] * GRW + coords[2 * n + 1]] = n;
}
__global__ void dwconv_kernel(const float* __restrict__ x, const float* __restrict__ ck,
                              const float* __restrict__ cb, const int* __restrict__ map,
                              const int* __restrict__ coords, float* __restrict__ y)
{
  const int idx = blockIdx.x * blockDim.x + threadIdx.x;
  if (idx >= N_TOK * DIM) return;
  const int n = idx / DIM, d = idx % DIM;
  const int r = coords[2 * n], c = coords[2 * n + 1];
  const float* w = ck + (size_t)d * 9;
  float acc = 0.0f;
#pragma unroll
  for (int dy = 0; dy < 3; ++dy) {
#pragma unroll
    for (int dx = 0; dx < 3; ++dx) {
      const int rr = r + dy - 1, cc = c + dx - 1;
      if (rr >= 0 && rr < GRH && cc >= 0 && cc < GRW) {
        const int m = map[rr * GRW + cc];
        if (m >= 0) acc += w[dy * 3 + dx] * x[(size_t)m * DIM + d];
      }
    }
  }
  y[(size_t)n * DIM + d] = x[(size_t)n * DIM + d] + acc + cb[d];
}

// ---------------- f32 -> f16 conversion ----------------
__global__ void f32_to_f16_kernel(const float* __restrict__ in, _Float16* __restrict__ out, int n) {
  const int idx = blockIdx.x * blockDim.x + threadIdx.x;
  if (idx < n) out[idx] = (_Float16)in[idx];
}

// ======================= host orchestration =======================
static inline size_t align_up(size_t x, size_t a) { return (x + a - 1) & ~(a - 1); }

extern "C" void kernel_launch(void* const* d_in, const int* in_sizes, int n_in,
                              void* d_out, int out_size, void* d_ws, size_t ws_size,
                              hipStream_t stream) {
  (void)in_sizes; (void)n_in; (void)out_size; (void)ws_size;

  const float* global_feats = (const float*)d_in[0];
  const int*   coords       = (const int*)  d_in[1];
  const float* ln1_g = (const float*)d_in[4];
  const float* ln1_b = (const float*)d_in[5];
  const float* wqkv  = (const float*)d_in[6];
  const float* bqkv  = (const float*)d_in[7];
  const float* wo    = (const float*)d_in[8];
  const float* bo    = (const float*)d_in[9];
  const float* ln2_g = (const float*)d_in[10];
  const float* ln2_b = (const float*)d_in[11];
  const float* w1    = (const float*)d_in[12];
  const float* b1    = (const float*)d_in[13];
  const float* w2    = (const float*)d_in[14];
  const float* b2    = (const float*)d_in[15];
  const float* ck    = (const float*)d_in[16];
  const float* cb    = (const float*)d_in[17];
  const float* lnf_g = (const float*)d_in[18];
  const float* lnf_b = (const float*)d_in[19];
  const float* wp    = (const float*)d_in[20];
  const float* bp    = (const float*)d_in[21];

  // ---- workspace layout ----
  char* ws = (char*)d_ws;
  size_t off = 0;
  auto alloc = [&](size_t bytes) { size_t o = off; off = align_up(off + bytes, 256); return (void*)(ws + o); };

  float*    xA     = (float*)   alloc((size_t)N_TOK * DIM * 4);
  float*    xB     = (float*)   alloc((size_t)N_TOK * DIM * 4);
  _Float16* xn16   = (_Float16*)alloc((size_t)N_TOK * DIM * 2);
  _Float16* qkv16  = (_Float16*)alloc((size_t)N_TOK * 3 * DIM * 2);
  _Float16* Qh     = (_Float16*)alloc((size_t)NHEAD * N_TOK * DHEAD * 2);
  _Float16* Kh     = (_Float16*)alloc((size_t)NHEAD * N_TOK * DHEAD * 2);
  _Float16* Vt     = (_Float16*)alloc((size_t)NHEAD * DHEAD * N_TOK * 2);
  _Float16* ao16   = (_Float16*)alloc((size_t)N_TOK * DIM * 2);
  _Float16* h16    = (_Float16*)alloc((size_t)N_TOK * FF * 2);
  _Float16* wqkv16 = (_Float16*)alloc((size_t)NDEPTH * 3 * DIM * DIM * 2);
  _Float16* wo16   = (_Float16*)alloc((size_t)NDEPTH * DIM * DIM * 2);
  _Float16* w1_16  = (_Float16*)alloc((size_t)NDEPTH * FF * DIM * 2);
  _Float16* w2_16  = (_Float16*)alloc((size_t)NDEPTH * DIM * FF * 2);
  _Float16* wp16   = (_Float16*)alloc((size_t)NGENES * DIM * 2);
  int*      map    = (int*)     alloc((size_t)GRH * GRW * 4);

  // ---- convert weights to f16 ----
  auto cvt = [&](const float* src, _Float16* dst, int n) {
    f32_to_f16_kernel<<<(n + 255) / 256, 256, 0, stream>>>(src, dst, n);
  };
  cvt(wqkv, wqkv16, NDEPTH * 3 * DIM * DIM);
  cvt(wo,   wo16,   NDEPTH * DIM * DIM);
  cvt(w1,   w1_16,  NDEPTH * FF * DIM);
  cvt(w2,   w2_16,  NDEPTH * DIM * FF);
  cvt(wp,   wp16,   NGENES * DIM);

  // ---- initial x, grid map ----
  hipMemcpyAsync(xA, global_feats, (size_t)N_TOK * DIM * 4, hipMemcpyDeviceToDevice, stream);
  grid_init_kernel<<<(GRH * GRW + 255) / 256, 256, 0, stream>>>(map);
  grid_scatter_kernel<<<(N_TOK + 255) / 256, 256, 0, stream>>>(coords, map);

  float* xcur = xA;
  float* xalt = xB;
  const int ND = N_TOK * DIM;
  const dim3 gemm_blk(128);

  for (int i = 0; i < NDEPTH; ++i) {
    // LN1 -> xn16
    layernorm_kernel<<<N_TOK, 256, 0, stream>>>(xcur, ln1_g + i * DIM, ln1_b + i * DIM, xn16, nullptr);
    // QKV GEMM: (4096 x 1536) = xn16 @ wqkv^T
    gemm_wmma_kernel<<<dim3(3 * DIM / 64, N_TOK / 128), gemm_blk, 0, stream>>>(
        xn16, wqkv16 + (size_t)i * 3 * DIM * DIM, bqkv + (size_t)i * 3 * DIM,
        nullptr, nullptr, qkv16, N_TOK, 3 * DIM, DIM, 0);
    qkv_pack_kernel<<<(ND + 255) / 256, 256, 0, stream>>>(qkv16, Qh, Kh, Vt);
    // attention -> ao16
    attention_kernel<<<dim3(N_TOK / 64, NHEAD), 128, 0, stream>>>(Qh, Kh, Vt, ao16);
    // proj + residual: x += ao @ wo^T + bo
    gemm_wmma_kernel<<<dim3(DIM / 64, N_TOK / 128), gemm_blk, 0, stream>>>(
        ao16, wo16 + (size_t)i * DIM * DIM, bo + (size_t)i * DIM,
        xcur, xcur, nullptr, N_TOK, DIM, DIM, 0);
    // LN2 -> xn16
    layernorm_kernel<<<N_TOK, 256, 0, stream>>>(xcur, ln2_g + i * DIM, ln2_b + i * DIM, xn16, nullptr);
    // MLP up + GELU -> h16
    gemm_wmma_kernel<<<dim3(FF / 64, N_TOK / 128), gemm_blk, 0, stream>>>(
        xn16, w1_16 + (size_t)i * FF * DIM, b1 + (size_t)i * FF,
        nullptr, nullptr, h16, N_TOK, FF, DIM, 1);
    // MLP down + residual
    gemm_wmma_kernel<<<dim3(DIM / 64, N_TOK / 128), gemm_blk, 0, stream>>>(
        h16, w2_16 + (size_t)i * DIM * FF, b2 + (size_t)i * DIM,
        xcur, xcur, nullptr, N_TOK, DIM, FF, 0);
    // depthwise conv + residual -> xalt; swap
    dwconv_kernel<<<(ND + 255) / 256, 256, 0, stream>>>(
        xcur, ck + (size_t)i * DIM * 9, cb + (size_t)i * DIM, map, coords, xalt);
    float* t = xcur; xcur = xalt; xalt = t;
  }

  // final LN: f32 into d_out (first output), f16 for the projection GEMM
  float* out_x    = (float*)d_out;
  float* out_pred = out_x + (size_t)N_TOK * DIM;
  layernorm_kernel<<<N_TOK, 256, 0, stream>>>(xcur, lnf_g, lnf_b, xn16, out_x);
  // pred = xn @ wp^T + bp  (250 columns; tail tiles guarded)
  gemm_wmma_kernel<<<dim3((NGENES + 63) / 64, N_TOK / 128), gemm_blk, 0, stream>>>(
      xn16, wp16, bp, nullptr, out_pred, nullptr, N_TOK, NGENES, DIM, 0);
}
